// pGNNNet_1_64441689309210
// MI455X (gfx1250) — matrix operand order repaired and checked by reading, not compiled, not run on hardware
//
#include <hip/hip_runtime.h>
#include <hip/hip_bf16.h>
#include <stddef.h>

// ---------------------------------------------------------------------------
// pGNN on MI455X (gfx1250): bf16 WMMA GEMMs + L2-resident atomic propagation
// ---------------------------------------------------------------------------

typedef __bf16  v16bf __attribute__((ext_vector_type(16)));
typedef float   v8f   __attribute__((ext_vector_type(8)));

#define IN_CH   512
#define HID_CH  256
#define OUT_CH  48
#define KSTEPS  4
#define COEF    0.08f          /* 2*mu/p = 2*0.1/2.5 */
#define EPSV    1e-12f

static __device__ __forceinline__ __bf16 f2bf(float f) {
    unsigned u = __float_as_uint(f);
    u += 0x7FFFu + ((u >> 16) & 1u);          // round-to-nearest-even
    unsigned short h = (unsigned short)(u >> 16);
    __bf16 r;
    __builtin_memcpy(&r, &h, sizeof(r));
    return r;
}

// ---------- weight transpose + fp32->bf16 conversion (one-time, tiny) -------
// wt[k*NOUT + n] = bf16( w[n*KIN + k] )
__global__ void cvt_weight_kernel(const float* __restrict__ w,
                                  __bf16* __restrict__ wt,
                                  int KIN, int NOUT) {
    int idx = blockIdx.x * blockDim.x + threadIdx.x;
    if (idx >= KIN * NOUT) return;
    int k = idx / NOUT;
    int n = idx - k * NOUT;
    wt[idx] = f2bf(w[(size_t)n * KIN + k]);
}

// ---------- bf16 WMMA GEMM: D = act(A[M,KC](f32) x Bt[KC,NC](bf16) + bias) --
// One wave computes a 16(M) x NT*16(N) tile. n-range split via blockIdx.y.
template <int KC, int NC, int NT, bool RELU>
__global__ __launch_bounds__(128)
void gemm_bf16_wmma(const float* __restrict__ A,
                    const __bf16* __restrict__ Bt,
                    const float* __restrict__ bias,
                    float* __restrict__ D, int M) {
    const int wave = blockIdx.x * (blockDim.x >> 5) + (threadIdx.x >> 5);
    const int lane = threadIdx.x & 31;
    const int m0   = wave * 16;
    if (m0 >= M) return;                       // wave-uniform: EXEC stays full
    const int n0 = blockIdx.y * (NT * 16);

    v8f acc[NT];
#pragma unroll
    for (int t = 0; t < NT; ++t) acc[t] = v8f{0.f,0.f,0.f,0.f,0.f,0.f,0.f,0.f};

    const int ml    = lane & 15;               // M row within tile
    const int khalf = (lane >> 4) * 8;         // lanes 16..31 take K+8 halves
    const float* arow = A + (size_t)(m0 + ml) * KC + khalf;

    for (int k0 = 0; k0 < KC; k0 += 32) {
        // A fragment per ISA layout: halves 0..7 = K kb..kb+7, 8..15 = kb+16..kb+23
        float4 fa = *(const float4*)(arow + k0);
        float4 fb = *(const float4*)(arow + k0 + 4);
        float4 fc = *(const float4*)(arow + k0 + 16);
        float4 fd = *(const float4*)(arow + k0 + 20);
        v16bf a;
        a[0]=f2bf(fa.x); a[1]=f2bf(fa.y); a[2]=f2bf(fa.z); a[3]=f2bf(fa.w);
        a[4]=f2bf(fb.x); a[5]=f2bf(fb.y); a[6]=f2bf(fb.z); a[7]=f2bf(fb.w);
        a[8]=f2bf(fc.x); a[9]=f2bf(fc.y); a[10]=f2bf(fc.z); a[11]=f2bf(fc.w);
        a[12]=f2bf(fd.x); a[13]=f2bf(fd.y); a[14]=f2bf(fd.z); a[15]=f2bf(fd.w);

        const __bf16* brow = Bt + (size_t)(k0 + khalf) * NC + n0 + ml;  // lane%16 = N col, K in table order
#pragma unroll
        for (int t = 0; t < NT; ++t) {
            v16bf b;
#pragma unroll
            for (int s = 0; s < 8; ++s) {
                b[s]     = brow[(size_t)s * NC + t * 16];
                b[s + 8] = brow[(size_t)(16 + s) * NC + t * 16];
            }
            acc[t] = __builtin_amdgcn_wmma_f32_16x16x32_bf16(
                         false, a, false, b, (short)0, acc[t], false, false);
        }
    }

    const int mb = (lane >> 4) * 8;            // C layout: lanes>=16 hold M+8
#pragma unroll
    for (int t = 0; t < NT; ++t) {
        const int n  = n0 + t * 16 + ml;
        const float bv = bias[n];
#pragma unroll
        for (int j = 0; j < 8; ++j) {
            float v = acc[t][j] + bv;
            if (RELU) v = fmaxf(v, 0.f);
            D[(size_t)(m0 + mb + j) * NC + n] = v;
        }
    }
}

// ---------- degree / normalization ------------------------------------------
__global__ void deg_init_kernel(float* __restrict__ deg, int N) {
    int i = blockIdx.x * blockDim.x + threadIdx.x;
    if (i < N) deg[i] = 1.0f;                  // self-loop
}
__global__ void deg_accum_kernel(const int* __restrict__ row,
                                 float* __restrict__ deg, int E) {
    int e = blockIdx.x * blockDim.x + threadIdx.x;
    if (e < E) atomicAdd(deg + row[e], 1.0f);
}
__global__ void dinv_kernel(float* __restrict__ deg, int N) {
    int i = blockIdx.x * blockDim.x + threadIdx.x;
    if (i < N) deg[i] = rsqrtf(deg[i]);        // deg >= 1 always
}

// ---------- propagation step -------------------------------------------------
__global__ void prop_init_kernel(float* __restrict__ denom,
                                 float* __restrict__ agg, int N) {
    int idx = blockIdx.x * blockDim.x + threadIdx.x;
    if (idx < N * OUT_CH) agg[idx] = 0.f;
    if (idx < N) denom[idx] = COEF;
}

// 16 lanes per edge, 3 features/lane. E real edges + N self-loops.
__global__ void edge_kernel(const int* __restrict__ row,
                            const int* __restrict__ col,
                            const float* __restrict__ dinv,
                            const float* __restrict__ out,
                            float* __restrict__ denom,
                            float* __restrict__ agg,
                            int E, int total) {
    int tid = blockIdx.x * blockDim.x + threadIdx.x;
    int e = tid >> 4, l = tid & 15;
    if (e >= total) return;
    int r, c;
    if (e < E) { r = row[e]; c = col[e]; } else { r = e - E; c = r; }
    const float w = dinv[r] * dinv[c];
    const float* orow = out + (size_t)r * OUT_CH;
    const float* ocol = out + (size_t)c * OUT_CH;
    float c0 = ocol[l], c1 = ocol[l + 16], c2 = ocol[l + 32];
    float d0 = orow[l] - c0, d1 = orow[l + 16] - c1, d2 = orow[l + 32] - c2;
    float ssq = d0 * d0 + d1 * d1 + d2 * d2;
    ssq += __shfl_xor(ssq, 1);
    ssq += __shfl_xor(ssq, 2);
    ssq += __shfl_xor(ssq, 4);
    ssq += __shfl_xor(ssq, 8);                 // stays inside each 16-lane group
    float gnorm = sqrtf(w * ssq + EPSV);
    float Mv = w * sqrtf(gnorm);               // gnorm^(p-2), p=2.5
    if (l == 0) atomicAdd(denom + r, Mv);
    float* ar = agg + (size_t)r * OUT_CH;
    atomicAdd(ar + l,      Mv * c0);
    atomicAdd(ar + l + 16, Mv * c1);
    atomicAdd(ar + l + 32, Mv * c2);
}

__global__ void node_update_kernel(const float* __restrict__ agg,
                                   const float* __restrict__ denom,
                                   const float* __restrict__ f0,
                                   float* __restrict__ outn, int total) {
    int idx = blockIdx.x * blockDim.x + threadIdx.x;
    if (idx >= total) return;
    int i = idx / OUT_CH;
    float alpha = 1.0f / denom[i];             // denom pre-seeded with COEF
    outn[idx] = alpha * agg[idx] + (COEF * alpha) * f0[idx];
}

// ---------- log-softmax over 48 features (16 lanes x 3 per node) -------------
__global__ void log_softmax_kernel(const float* __restrict__ in,
                                   float* __restrict__ out, int N) {
    int tid = blockIdx.x * blockDim.x + threadIdx.x;
    int node = tid >> 4, l = tid & 15;
    if (node >= N) return;
    const float* p = in + (size_t)node * OUT_CH;
    float v0 = p[l], v1 = p[l + 16], v2 = p[l + 32];
    float m = fmaxf(v0, fmaxf(v1, v2));
    m = fmaxf(m, __shfl_xor(m, 1));
    m = fmaxf(m, __shfl_xor(m, 2));
    m = fmaxf(m, __shfl_xor(m, 4));
    m = fmaxf(m, __shfl_xor(m, 8));
    float s = __expf(v0 - m) + __expf(v1 - m) + __expf(v2 - m);
    s += __shfl_xor(s, 1);
    s += __shfl_xor(s, 2);
    s += __shfl_xor(s, 4);
    s += __shfl_xor(s, 8);
    float ls = m + __logf(s);
    float* q = out + (size_t)node * OUT_CH;
    q[l] = v0 - ls; q[l + 16] = v1 - ls; q[l + 32] = v2 - ls;
}

// ---------------------------------------------------------------------------
extern "C" void kernel_launch(void* const* d_in, const int* in_sizes, int n_in,
                              void* d_out, int out_size, void* d_ws, size_t ws_size,
                              hipStream_t stream) {
    const float* x   = (const float*)d_in[0];
    const float* w1  = (const float*)d_in[1];
    const float* b1  = (const float*)d_in[2];
    const float* w2  = (const float*)d_in[3];
    const float* b2  = (const float*)d_in[4];
    const int*   ei  = (const int*)d_in[5];

    const int N = in_sizes[0] / IN_CH;         // 50000
    const int E = in_sizes[5] / 2;             // 1600000
    const int* row = ei;
    const int* col = ei + E;

    // workspace layout (256B-aligned offsets)
    char* ws = (char*)d_ws;
    __bf16* w1t = (__bf16*)(ws);                                   // 512*256*2
    __bf16* w2t = (__bf16*)(ws + 262144);                          // 256*48*2
    float*  h   = (float*)(ws + 286720);                           // N*256*4
    float*  f0  = (float*)(ws + 286720 + (size_t)N * HID_CH * 4);  // N*48*4
    float*  deg = (float*)((char*)f0 + (size_t)N * OUT_CH * 4);    // N*4 (-> dinv)
    // propagation buffers alias dead h region after GEMM2
    float* bufA  = h;
    float* bufB  = (float*)((char*)h +  9600000);
    float* agg   = (float*)((char*)h + 19200000);
    float* denom = (float*)((char*)h + 28800000);

    const int T = 256;

    // 1) weight transposes -> bf16 K-major
    cvt_weight_kernel<<<(IN_CH * HID_CH + T - 1) / T, T, 0, stream>>>(w1, w1t, IN_CH, HID_CH);
    cvt_weight_kernel<<<(HID_CH * OUT_CH + T - 1) / T, T, 0, stream>>>(w2, w2t, HID_CH, OUT_CH);

    // 2) h = relu(x @ w1^T + b1)   [N,512]x[512,256]
    {
        int waves = (N + 15) / 16;
        dim3 grid((waves + 3) / 4, HID_CH / (8 * 16));
        gemm_bf16_wmma<IN_CH, HID_CH, 8, true><<<grid, 128, 0, stream>>>(x, w1t, b1, h, N);
    }
    // 3) f0 = h @ w2^T + b2        [N,256]x[256,48]
    {
        int waves = (N + 15) / 16;
        dim3 grid((waves + 3) / 4, 1);
        gemm_bf16_wmma<HID_CH, OUT_CH, 3, false><<<grid, 128, 0, stream>>>(h, w2t, b2, f0, N);
    }

    // 4) symmetric GCN normalization: dinv = rsqrt(deg), deg includes self-loop
    deg_init_kernel<<<(N + T - 1) / T, T, 0, stream>>>(deg, N);
    deg_accum_kernel<<<(E + T - 1) / T, T, 0, stream>>>(row, deg, E);
    dinv_kernel<<<(N + T - 1) / T, T, 0, stream>>>(deg, N);

    // 5) K-step p-Laplacian propagation (ping-pong, f0 preserved)
    const int totalE   = E + N;
    const int edgeThr  = totalE * 16;
    const int nodeThr  = N * OUT_CH;
    const float* cur = f0;
    float* buffers[2] = { bufA, bufB };
    float* nxt = bufA;
    for (int step = 0; step < KSTEPS; ++step) {
        nxt = buffers[step & 1];
        prop_init_kernel<<<(nodeThr + T - 1) / T, T, 0, stream>>>(denom, agg, N);
        edge_kernel<<<(edgeThr + T - 1) / T, T, 0, stream>>>(row, col, deg, cur,
                                                             denom, agg, E, totalE);
        node_update_kernel<<<(nodeThr + T - 1) / T, T, 0, stream>>>(agg, denom, f0,
                                                                    nxt, nodeThr);
        cur = nxt;
    }

    // 6) log_softmax -> d_out
    log_softmax_kernel<<<(N * 16 + T - 1) / T, T, 0, stream>>>(cur, (float*)d_out, N);
}
